// MPNN_54640573939922
// MI455X (gfx1250) — matrix operand order repaired and checked by reading, not compiled, not run on hardware
//
#include <hip/hip_runtime.h>
#include <hip/hip_bf16.h>

typedef __attribute__((ext_vector_type(16))) __bf16 v16bf;
typedef __attribute__((ext_vector_type(8)))  float  v8f;

#define NN 20000
#define EE 640000
#define HH 128
#define LL 3
#define GG 128
#define CC 2
#define TM 5                 // row tiles (of 16) per block; 1250 % 5 == 0
#define RBTOT (NN / 16)      // 1250 row tiles

__device__ __forceinline__ __bf16 f2bf(float f) { return (__bf16)f; }

// C-fragment flat index for element (row, col) of an N x 128 matrix stored in
// WMMA D-layout order: [rblk][wave=col>>4][lane=(col&15)+16*((row>>3)&1)][i=row&7]
__device__ __forceinline__ size_t yfrag_idx(int row, int col) {
  return ((((size_t)(row >> 4) * 8 + (col >> 4)) * 32 + (col & 15) +
           ((row >> 3) & 1) * 16) * 8) + (row & 7);
}

// ---------------- encoder: h[n][j] = x[n]*enc_W[j] + enc_b[j] ----------------
__global__ __launch_bounds__(256) void gin_encoder(
    const float* __restrict__ x, const float* __restrict__ W,
    const float* __restrict__ b, float* __restrict__ h) {
  int t = blockIdx.x * 256 + threadIdx.x;
  if (t >= NN * HH) return;
  int n = t >> 7, j = t & 127;
  h[t] = x[n] * W[j] + b[j];
}

// ---------------- z = (1+eps)*h ----------------
__global__ __launch_bounds__(256) void gin_init_z(
    const float* __restrict__ h, const float* __restrict__ eps, int l,
    float* __restrict__ z) {
  int t = blockIdx.x * 256 + threadIdx.x;
  if (t >= NN * HH) return;
  z[t] = (1.0f + eps[l]) * h[t];
}

// ---------------- scatter-add: z[dst] += h[src], 32 threads/edge x 4 cols ----
__global__ __launch_bounds__(256) void gin_scatter(
    const float* __restrict__ h, const int* __restrict__ src,
    const int* __restrict__ dst, float* __restrict__ z) {
  long long t = (long long)blockIdx.x * 256 + threadIdx.x;
  int e = (int)(t >> 5);
  if (e >= EE) return;
  int j = ((int)t & 31) * 4;
  const float4 v = *(const float4*)(h + (size_t)src[e] * HH + j);
  float* zp = z + (size_t)dst[e] * HH + j;
  atomicAdd(zp + 0, v.x); atomicAdd(zp + 1, v.y);
  atomicAdd(zp + 2, v.z); atomicAdd(zp + 3, v.w);
}

// ---------- weight pre-swizzle into B-fragment order (all L*2 matrices) ------
// out[((mat*4+kt)*8+nt)*512 + lane*16 + e] = W[layer][K][ncol]
//   K = kt*32 + (e>>3)*16 + (lane>>4)*8 + (e&7) ; ncol = nt*16 + (lane&15)
__global__ __launch_bounds__(256) void gin_wswz(
    const float* __restrict__ W1, const float* __restrict__ W2,
    __bf16* __restrict__ Wz) {
  int t = blockIdx.x * 256 + threadIdx.x;
  if (t >= LL * 2 * HH * HH) return;
  int e = t & 15, lane = (t >> 4) & 31, nt = (t >> 9) & 7, kt = (t >> 12) & 3;
  int mat = t >> 14;
  int K = kt * 32 + ((e >> 3) * 16) + ((lane >> 4) * 8) + (e & 7);
  int nc = nt * 16 + (lane & 15);
  int layer = mat >> 1;
  const float* Wp = (mat & 1) ? W2 : W1;
  Wz[t] = f2bf(Wp[((size_t)layer * HH + K) * HH + nc]);
}

// ---------- A-fragment source mapping ----------------------------------------
// out[((rblk*4+kt)*32+lane)*16+e]: row=rblk*16+(lane&15),
// K = kt*32 + (e>>3)*16 + (lane>>4)*8 + (e&7)
__device__ __forceinline__ void afrag_src(int t, int& row, int& K) {
  int e = t & 15, lane = (t >> 4) & 31, kt = (t >> 9) & 3, rblk = t >> 11;
  row = rblk * 16 + (lane & 15);
  K = kt * 32 + ((e >> 3) * 16) + ((lane >> 4) * 8) + (e & 7);
}

// ---------- convert fp32 row-major -> A-fragment bf16 order ------------------
__global__ __launch_bounds__(256) void gin_aswz(
    const float* __restrict__ z, __bf16* __restrict__ A) {
  int t = blockIdx.x * 256 + threadIdx.x;
  if (t >= NN * HH) return;
  int row, K; afrag_src(t, row, K);
  A[t] = f2bf(z[(size_t)row * HH + K]);
}

// ---------- fused BN(stats precomputed)+ReLU: Y(frag f32) -> A-fragment bf16 -
__global__ __launch_bounds__(256) void gin_bn_swz(
    const float* __restrict__ yf, const float* __restrict__ mean,
    const float* __restrict__ rsig, const float* __restrict__ gma,
    const float* __restrict__ bta, __bf16* __restrict__ A) {
  int t = blockIdx.x * 256 + threadIdx.x;
  if (t >= NN * HH) return;
  int row, K; afrag_src(t, row, K);
  float v = yf[yfrag_idx(row, K)];
  v = gma[K] * (v - mean[K]) * rsig[K] + bta[K];
  A[t] = f2bf(fmaxf(v, 0.0f));
}

// ---------- outer BN+ReLU: Y(frag f32) -> fp32 h (row-major) -----------------
__global__ __launch_bounds__(256) void gin_bn_f32(
    const float* __restrict__ yf, const float* __restrict__ mean,
    const float* __restrict__ rsig, const float* __restrict__ gma,
    const float* __restrict__ bta, float* __restrict__ h) {
  int t = blockIdx.x * 256 + threadIdx.x;
  if (t >= NN * HH) return;
  int row = t >> 7, j = t & 127;
  float v = yf[yfrag_idx(row, j)];
  v = gma[j] * (v - mean[j]) * rsig[j] + bta[j];
  h[t] = fmaxf(v, 0.0f);
}

// ---------- BN column stats over fragment-ordered Y: one block per feature ---
__global__ __launch_bounds__(256) void gin_bn_stats(
    const float* __restrict__ yf, float* __restrict__ mean,
    float* __restrict__ rsig) {
  int j = blockIdx.x;
  int wv = j >> 4, l0 = j & 15;
  float s = 0.f, s2 = 0.f;
  for (int rb = threadIdx.x; rb < RBTOT; rb += 256) {
    const float* p0 = yf + (((size_t)rb * 8 + wv) * 32 + l0) * 8;
    const float* p1 = yf + (((size_t)rb * 8 + wv) * 32 + l0 + 16) * 8;
#pragma unroll
    for (int i = 0; i < 8; ++i) { float v = p0[i]; s += v; s2 += v * v; }
#pragma unroll
    for (int i = 0; i < 8; ++i) { float v = p1[i]; s += v; s2 += v * v; }
  }
  __shared__ float sh[256], sh2[256];
  sh[threadIdx.x] = s; sh2[threadIdx.x] = s2; __syncthreads();
  for (int off = 128; off; off >>= 1) {
    if (threadIdx.x < off) {
      sh[threadIdx.x] += sh[threadIdx.x + off];
      sh2[threadIdx.x] += sh2[threadIdx.x + off];
    }
    __syncthreads();
  }
  if (threadIdx.x == 0) {
    float mu = sh[0] / (float)NN;
    float var = sh2[0] / (float)NN - mu * mu;
    mean[j] = mu;
    rsig[j] = rsqrtf(var + 1e-5f);
  }
}

// ---------- WMMA GEMM: Yf = A(bf16 frags) @ Wz(bf16 frags) + bias ------------
// block = 256 threads (8 waves); block owns TM*16 rows staged through LDS;
// wave w owns cols [16w,16w+16); B fragments held in registers across TM tiles.
__global__ __launch_bounds__(256) void gin_gemm_wmma(
    const __bf16* __restrict__ A, const __bf16* __restrict__ Wz,
    const float* __restrict__ bias, float* __restrict__ Yf) {
  __shared__ __attribute__((aligned(16))) __bf16 As[TM * 4 * 512];  // 20 KB
  int tid = threadIdx.x;
  int wave = tid >> 5, lane = tid & 31;

  // cooperative stage of TM A-tiles: TM*4*512 bf16 = 20480 B = 1280 x 16B
  size_t abase = (size_t)blockIdx.x * (TM * 4 * 512);
#pragma unroll
  for (int i = 0; i < (TM * 4 * 512) / (256 * 8); ++i) {
    int idx = (i * 256 + tid) * 8;
    *(float4*)(&As[idx]) = *(const float4*)(A + abase + idx);
  }
  __syncthreads();

  // B fragments for this wave's column tile, reused across all TM row tiles
  const __bf16* Bp = Wz + (size_t)wave * 512 + lane * 16;
  v16bf b0 = *(const v16bf*)(Bp + 0 * 8 * 512);
  v16bf b1 = *(const v16bf*)(Bp + 1 * 8 * 512);
  v16bf b2 = *(const v16bf*)(Bp + 2 * 8 * 512);
  v16bf b3 = *(const v16bf*)(Bp + 3 * 8 * 512);

  int col = wave * 16 + (lane & 15);
  float bv = bias[col];

#pragma unroll
  for (int rt = 0; rt < TM; ++rt) {
    v8f c = {bv, bv, bv, bv, bv, bv, bv, bv};
    const __bf16* Ap = &As[(rt * 4) * 512 + lane * 16];
    v16bf a;
    a = *(const v16bf*)(Ap + 0 * 512);
    c = __builtin_amdgcn_wmma_f32_16x16x32_bf16(false, a, false, b0, (short)0,
                                                c, false, false);
    a = *(const v16bf*)(Ap + 1 * 512);
    c = __builtin_amdgcn_wmma_f32_16x16x32_bf16(false, a, false, b1, (short)0,
                                                c, false, false);
    a = *(const v16bf*)(Ap + 2 * 512);
    c = __builtin_amdgcn_wmma_f32_16x16x32_bf16(false, a, false, b2, (short)0,
                                                c, false, false);
    a = *(const v16bf*)(Ap + 3 * 512);
    c = __builtin_amdgcn_wmma_f32_16x16x32_bf16(false, a, false, b3, (short)0,
                                                c, false, false);
    // dense C-fragment store: two b128 per lane
    size_t o = (((size_t)(blockIdx.x * TM + rt) * 8 + wave) * 32 + lane) * 8;
    *(float4*)(Yf + o)     = make_float4(c[0], c[1], c[2], c[3]);
    *(float4*)(Yf + o + 4) = make_float4(c[4], c[5], c[6], c[7]);
  }
}

// ---------- pooling ----------------------------------------------------------
__global__ __launch_bounds__(256) void gin_pool_zero(float* __restrict__ pooled,
                                                     float* __restrict__ cnt) {
  int t = blockIdx.x * 256 + threadIdx.x;
  if (t < GG * HH) pooled[t] = 0.f;
  if (t < GG) cnt[t] = 0.f;
}

__global__ __launch_bounds__(256) void gin_pool_scatter(
    const float* __restrict__ h, const int* __restrict__ batch,
    float* __restrict__ pooled) {
  long long t = (long long)blockIdx.x * 256 + threadIdx.x;
  int n = (int)(t >> 5);
  if (n >= NN) return;
  int j = ((int)t & 31) * 4;
  const float4 v = *(const float4*)(h + (size_t)n * HH + j);
  float* p = pooled + (size_t)batch[n] * HH + j;
  atomicAdd(p + 0, v.x); atomicAdd(p + 1, v.y);
  atomicAdd(p + 2, v.z); atomicAdd(p + 3, v.w);
}

__global__ __launch_bounds__(256) void gin_pool_count(
    const int* __restrict__ batch, float* __restrict__ cnt) {
  int n = blockIdx.x * 256 + threadIdx.x;
  if (n >= NN) return;
  atomicAdd(&cnt[batch[n]], 1.0f);
}

__global__ __launch_bounds__(256) void gin_classifier(
    const float* __restrict__ pooled, const float* __restrict__ cnt,
    const float* __restrict__ cW, const float* __restrict__ cb,
    float* __restrict__ out) {
  int t = threadIdx.x;  // one block, 256 threads = G*C
  if (t >= GG * CC) return;
  int g = t / CC, c = t % CC;
  float s = 0.f;
  for (int j = 0; j < HH; ++j) s += pooled[(size_t)g * HH + j] * cW[j * CC + c];
  out[t] = s / fmaxf(cnt[g], 1.0f) + cb[c];
}

extern "C" void kernel_launch(void* const* d_in, const int* in_sizes, int n_in,
                              void* d_out, int out_size, void* d_ws,
                              size_t ws_size, hipStream_t stream) {
  const float* x     = (const float*)d_in[0];
  const int*   eidx  = (const int*)d_in[1];
  const int*   batch = (const int*)d_in[2];
  const float* encW  = (const float*)d_in[3];
  const float* encb  = (const float*)d_in[4];
  const float* W1    = (const float*)d_in[5];
  const float* b1    = (const float*)d_in[6];
  const float* g1    = (const float*)d_in[7];
  const float* be1   = (const float*)d_in[8];
  const float* W2    = (const float*)d_in[9];
  const float* b2    = (const float*)d_in[10];
  const float* eps   = (const float*)d_in[11];
  const float* bng   = (const float*)d_in[12];
  const float* bnb   = (const float*)d_in[13];
  const float* cW    = (const float*)d_in[14];
  const float* cb    = (const float*)d_in[15];
  float* out = (float*)d_out;

  const int* esrc = eidx;
  const int* edst = eidx + EE;

  // workspace carve-out (all offsets 256B aligned)
  char* ws = (char*)d_ws;
  size_t off = 0;
  auto take = [&](size_t bytes) {
    char* p = ws + off;
    off += (bytes + 255) & ~(size_t)255;
    return p;
  };
  float*  h      = (float*)take((size_t)NN * HH * 4);   // row-major features
  float*  z      = (float*)take((size_t)NN * HH * 4);   // row-major pre-GEMM
  float*  yf     = (float*)take((size_t)NN * HH * 4);   // fragment-ordered GEMM out
  float*  zf     = (float*)take((size_t)NN * HH * 4);   // fragment-ordered GEMM out
  __bf16* Abuf   = (__bf16*)take((size_t)NN * HH * 2);
  __bf16* Wz     = (__bf16*)take((size_t)LL * 2 * HH * HH * 2);
  float*  mean   = (float*)take(HH * 4);
  float*  rsig   = (float*)take(HH * 4);
  float*  pooled = (float*)take((size_t)GG * HH * 4);
  float*  cnt    = (float*)take(GG * 4);
  (void)ws_size;

  const int NH_BLK = (NN * HH + 255) / 256;                    // 10000
  const int SC_BLK = (int)(((long long)EE * 32 + 255) / 256);  // 80000
  const int GEMM_BLK = RBTOT / TM;                             // 250

  gin_wswz<<<(LL * 2 * HH * HH + 255) / 256, 256, 0, stream>>>(W1, W2, Wz);
  gin_encoder<<<NH_BLK, 256, 0, stream>>>(x, encW, encb, h);

  for (int l = 0; l < LL; ++l) {
    gin_init_z<<<NH_BLK, 256, 0, stream>>>(h, eps, l, z);
    gin_scatter<<<SC_BLK, 256, 0, stream>>>(h, esrc, edst, z);
    gin_aswz<<<NH_BLK, 256, 0, stream>>>(z, Abuf);
    gin_gemm_wmma<<<GEMM_BLK, 256, 0, stream>>>(
        Abuf, Wz + (size_t)(l * 2) * HH * HH, b1 + l * HH, yf);
    gin_bn_stats<<<HH, 256, 0, stream>>>(yf, mean, rsig);
    gin_bn_swz<<<NH_BLK, 256, 0, stream>>>(yf, mean, rsig, g1 + l * HH,
                                           be1 + l * HH, Abuf);
    gin_gemm_wmma<<<GEMM_BLK, 256, 0, stream>>>(
        Abuf, Wz + (size_t)(l * 2 + 1) * HH * HH, b2 + l * HH, zf);
    gin_bn_stats<<<HH, 256, 0, stream>>>(zf, mean, rsig);
    gin_bn_f32<<<NH_BLK, 256, 0, stream>>>(zf, mean, rsig, bng + l * HH,
                                           bnb + l * HH, h);
  }

  gin_pool_zero<<<(GG * HH + 255) / 256, 256, 0, stream>>>(pooled, cnt);
  gin_pool_scatter<<<(NN * 32 + 255) / 256, 256, 0, stream>>>(h, batch, pooled);
  gin_pool_count<<<(NN + 255) / 256, 256, 0, stream>>>(batch, cnt);
  gin_classifier<<<1, 256, 0, stream>>>(pooled, cnt, cW, cb, out);
}
